// MambaFull_28681791602700
// MI455X (gfx1250) — compile-verified, matrix operand order checked by reading, not run on hardware
//
#include <hip/hip_runtime.h>
#include <hip/hip_bf16.h>

// ---------------------------------------------------------------------------
// Mamba TSP forward for gfx1250 (MI455X): bf16 WMMA GEMMs + VALU scan.
// B=32, L=300 (T=9600 tokens), d_model=256, d_inner=512, d_state=16,
// dt_rank=16, conv=4, layers=4.  T is a multiple of 64 -> no M guards.
// ---------------------------------------------------------------------------

#define T_TOK   9600
#define DMODEL  256
#define DINNER  512
#define DSTATE  16
#define DTRANK  16
#define SEQ     300
#define NLAYERS 4

typedef __attribute__((ext_vector_type(16))) __bf16 v16bf;
typedef __attribute__((ext_vector_type(8)))  float  v8f;

// Native bf16 convert (fptrunc, RNE) -> v_cvt_*bf16 on gfx1250; pairs fuse.
__device__ __forceinline__ unsigned short f2bf(float f) {
    union { __bf16 h; unsigned short u; } v;
    v.h = (__bf16)f;
    return v.u;
}

__device__ __forceinline__ unsigned pk2(float lo, float hi) {
    union { __bf16 h[2]; unsigned u; } v;
    v.h[0] = (__bf16)lo;
    v.h[1] = (__bf16)hi;
    return v.u;
}

__device__ __forceinline__ float silu_f(float x) {
    return x / (1.0f + __expf(-x));
}

__device__ __forceinline__ float softplus_f(float x) {
    return (x > 20.0f) ? x : __logf(1.0f + __expf(x));
}

// ---------------------------------------------------------------------------
// GEMM: C[M,N] = epi( A[M,K] * W[N,K]^T ), fp32 in/out, bf16 WMMA compute.
// Block = 256 threads = 8 waves -> 64(M) x 64(N) tile. Each wave: 32x16
// output (two v8f accumulators sharing one B fragment), K-loop step 32.
// Requires M % 64 == 0 (true for all call sites: M = 9600).
// FULLK  : K % 32 == 0 -> unconditional float4 staging loads.
// FULLN  : N % 64 == 0 -> unguarded stores; else clamp B rows + guard stores.
// BIASACT: fused bias + softplus epilogue (dt_proj only).
// ---------------------------------------------------------------------------
#define TM 64
#define TN 64
#define TK 32

template<bool FULLK, bool FULLN, bool BIASACT>
__global__ __launch_bounds__(256) void wmma_gemm_bf16(
    const float* __restrict__ A, int lda,
    const float* __restrict__ W, int ldw,
    float* __restrict__ C, int ldc,
    const float* __restrict__ bias,
    int N, int K)
{
    __shared__ __align__(16) unsigned short As[TM][TK + 8];   // 80B row stride
    __shared__ __align__(16) unsigned short Bs[TN][TK + 8];

    const int tid  = threadIdx.x;
    const int wave = tid >> 5;
    const int lane = tid & 31;
    const int hf   = lane >> 4;
    const int l16  = lane & 15;
    const int wm   = wave & 1;      // 2 M-super-tiles of 32 rows
    const int wn   = wave >> 1;     // 4 N-tiles of 16 cols

    const int mBase = blockIdx.y * TM;
    const int nBase = blockIdx.x * TN;

    v8f acc0 = {};
    v8f acc1 = {};

    // Staging geometry: 256 threads cover a 64x32 tile, 8 K-values each.
    const int sRow = tid >> 2;                 // 0..63
    const int sK0  = (tid & 3) * 8;            // 0,8,16,24
    const int aRow = mBase + sRow;             // always < M
    const int bRowRaw = nBase + sRow;
    const int bRow = FULLN ? bRowRaw
                           : (bRowRaw < N - 1 ? bRowRaw : N - 1);

    for (int kb = 0; kb < K; kb += TK) {
        if (FULLK) {
            const float4* ap = (const float4*)(A + (size_t)aRow * lda + kb + sK0);
            const float4* bp = (const float4*)(W + (size_t)bRow * ldw + kb + sK0);
            const float4 a0 = ap[0], a1 = ap[1];
            const float4 b0 = bp[0], b1 = bp[1];
            uint4 pa, pb;
            pa.x = pk2(a0.x, a0.y); pa.y = pk2(a0.z, a0.w);
            pa.z = pk2(a1.x, a1.y); pa.w = pk2(a1.z, a1.w);
            pb.x = pk2(b0.x, b0.y); pb.y = pk2(b0.z, b0.w);
            pb.z = pk2(b1.x, b1.y); pb.w = pk2(b1.z, b1.w);
            *(uint4*)&As[sRow][sK0] = pa;
            *(uint4*)&Bs[sRow][sK0] = pb;
        } else {
            // Ragged K (dt_proj, K=16): clamp address, select zero. Branchless.
#pragma unroll
            for (int j = 0; j < 8; ++j) {
                const int kk  = kb + sK0 + j;
                const int kks = kk < K - 1 ? kk : K - 1;
                float va = A[(size_t)aRow * lda + kks];
                float vb = W[(size_t)bRow * ldw + kks];
                va = (kk < K) ? va : 0.0f;
                vb = (kk < K) ? vb : 0.0f;
                As[sRow][sK0 + j] = f2bf(va);
                Bs[sRow][sK0 + j] = f2bf(vb);
            }
        }
        __syncthreads();

        // Fragment gather per ISA 7.12.2 (16-bit A 16x32 layout; B mirrored):
        // elements 0..7 -> K = hf*8 + 0..7 ; elements 8..15 -> K = 16+hf*8+..
        const int ar0 = wm * 32 + l16;
        const int ar1 = ar0 + 16;
        const int br  = wn * 16 + l16;
        v16bf af0, af1, bfr;
        {
            const __bf16* p0 = (const __bf16*)&As[ar0][hf * 8];
            const __bf16* p1 = (const __bf16*)&As[ar0][16 + hf * 8];
            const __bf16* p2 = (const __bf16*)&As[ar1][hf * 8];
            const __bf16* p3 = (const __bf16*)&As[ar1][16 + hf * 8];
            const __bf16* q0 = (const __bf16*)&Bs[br][hf * 8];
            const __bf16* q1 = (const __bf16*)&Bs[br][16 + hf * 8];
#pragma unroll
            for (int e = 0; e < 8; ++e) {
                af0[e] = p0[e]; af0[8 + e] = p1[e];
                af1[e] = p2[e]; af1[8 + e] = p3[e];
                bfr[e] = q0[e]; bfr[8 + e] = q1[e];
            }
        }

        acc0 = __builtin_amdgcn_wmma_f32_16x16x32_bf16(
                   false, af0, false, bfr, (short)0, acc0, false, false);
        acc1 = __builtin_amdgcn_wmma_f32_16x16x32_bf16(
                   false, af1, false, bfr, (short)0, acc1, false, false);
        __syncthreads();
    }

    // Epilogue: C/D layout -> VGPR r holds row (r + 8*hf), col = lane&15.
    const int col = nBase + wn * 16 + l16;
    float bv = 0.0f;
    if (BIASACT) bv = bias[FULLN ? col : (col < N - 1 ? col : N - 1)];
    if (FULLN || col < N) {
#pragma unroll
        for (int r = 0; r < 8; ++r) {
            const int row0 = mBase + wm * 32 + hf * 8 + r;
            float v0 = acc0[r];
            float v1 = acc1[r];
            if (BIASACT) {
                v0 = softplus_f(v0 + bv);
                v1 = softplus_f(v1 + bv);
            }
            C[(size_t)row0 * ldc + col] = v0;
            C[(size_t)(row0 + 16) * ldc + col] = v1;
        }
    }
}

// ---------------------------------------------------------------------------
// Embedding: x[g,f] = emb_W[f,0]*cx + emb_W[f,1]*cy + emb_b[f]
// ---------------------------------------------------------------------------
__global__ __launch_bounds__(256) void embed_kernel(
    const float* __restrict__ coords, const float* __restrict__ Wm,
    const float* __restrict__ bm, float* __restrict__ x, int total)
{
    const int idx = blockIdx.x * blockDim.x + threadIdx.x;
    if (idx >= total) return;
    const int f = idx & (DMODEL - 1);
    const int g = idx >> 8;
    x[idx] = Wm[f * 2] * coords[(size_t)g * 2] +
             Wm[f * 2 + 1] * coords[(size_t)g * 2 + 1] + bm[f];
}

// ---------------------------------------------------------------------------
// resid = (first ? x : x + resid); out = LayerNorm(resid). One token / block.
// ---------------------------------------------------------------------------
__global__ __launch_bounds__(256) void add_ln_kernel(
    const float* __restrict__ x, float* __restrict__ resid,
    const float* __restrict__ w, const float* __restrict__ b,
    float* __restrict__ out, int first)
{
    __shared__ float s1[256];
    __shared__ float s2[256];
    const int g = blockIdx.x;
    const int f = threadIdx.x;
    float v = x[(size_t)g * DMODEL + f];
    if (!first) v += resid[(size_t)g * DMODEL + f];
    resid[(size_t)g * DMODEL + f] = v;
    s1[f] = v; s2[f] = v * v;
    __syncthreads();
    for (int off = 128; off > 0; off >>= 1) {
        if (f < off) { s1[f] += s1[f + off]; s2[f] += s2[f + off]; }
        __syncthreads();
    }
    const float mean = s1[0] * (1.0f / DMODEL);
    const float var  = s2[0] * (1.0f / DMODEL) - mean * mean;
    const float inv  = rsqrtf(var + 1e-5f);
    out[(size_t)g * DMODEL + f] = (v - mean) * inv * w[f] + b[f];
}

// ---------------------------------------------------------------------------
// res = x + resid; out = RMSNorm(res) * w + b.
// ---------------------------------------------------------------------------
__global__ __launch_bounds__(256) void final_rms_kernel(
    const float* __restrict__ x, const float* __restrict__ resid,
    const float* __restrict__ w, const float* __restrict__ b,
    float* __restrict__ out)
{
    __shared__ float s2[256];
    const int g = blockIdx.x;
    const int f = threadIdx.x;
    const float v = x[(size_t)g * DMODEL + f] + resid[(size_t)g * DMODEL + f];
    s2[f] = v * v;
    __syncthreads();
    for (int off = 128; off > 0; off >>= 1) {
        if (f < off) s2[f] += s2[f + off];
        __syncthreads();
    }
    const float inv = rsqrtf(s2[0] * (1.0f / DMODEL) + 1e-5f);
    out[(size_t)g * DMODEL + f] = v * inv * w[f] + b[f];
}

// ---------------------------------------------------------------------------
// Causal depthwise conv1d (width 4) + SiLU. xi = xz[:, :, 0:512].
// Branchless: clamp the time index, select-zero the tap.
// ---------------------------------------------------------------------------
__global__ __launch_bounds__(256) void conv_silu_kernel(
    const float* __restrict__ xz, const float* __restrict__ cw,
    const float* __restrict__ cb, float* __restrict__ xc, int total)
{
    const int idx = blockIdx.x * blockDim.x + threadIdx.x;
    if (idx >= total) return;
    const int d = idx & (DINNER - 1);
    const int g = idx >> 9;
    const int bb = g / SEQ;
    const int t  = g - bb * SEQ;
    float acc = cb[d];
#pragma unroll
    for (int k = 0; k < 4; ++k) {
        const int tt  = t + k - 3;
        const int ttc = tt < 0 ? 0 : tt;
        const float v = xz[((size_t)(bb * SEQ + ttc)) * (2 * DINNER) + d];
        acc += (tt >= 0) ? cw[d * 4 + k] * v : 0.0f;
    }
    xc[idx] = silu_f(acc);
}

// ---------------------------------------------------------------------------
// Selective scan: one thread per (batch, channel). 16-state recurrence in
// registers, L=300 sequential steps, v_exp_f32 for exp(dt*A).
// dbl layout per token: [dt_lowrank(16) | B(16) | C(16)].
// ---------------------------------------------------------------------------
__global__ __launch_bounds__(256) void scan_kernel(
    const float* __restrict__ xc, const float* __restrict__ dt,
    const float* __restrict__ dbl, const float* __restrict__ A_log,
    float* __restrict__ y)
{
    const int idx = blockIdx.x * blockDim.x + threadIdx.x;
    if (idx >= 32 * DINNER) return;
    const int bb = idx >> 9;
    const int d  = idx & (DINNER - 1);

    float Arow[DSTATE], h[DSTATE];
#pragma unroll
    for (int s = 0; s < DSTATE; ++s) {
        Arow[s] = -__expf(A_log[d * DSTATE + s]);
        h[s] = 0.0f;
    }
    for (int t = 0; t < SEQ; ++t) {
        const size_t g = (size_t)bb * SEQ + t;
        const float xv  = xc[g * DINNER + d];
        const float dtv = dt[g * DINNER + d];
        const float* Bp = dbl + g * 48 + DTRANK;
        const float* Cp = dbl + g * 48 + DTRANK + DSTATE;
        const float dx = dtv * xv;
        float yv = 0.0f;
#pragma unroll
        for (int s = 0; s < DSTATE; ++s) {
            const float dA = __expf(dtv * Arow[s]);
            h[s] = dA * h[s] + dx * Bp[s];
            yv += h[s] * Cp[s];
        }
        y[g * DINNER + d] = yv;
    }
}

// ---------------------------------------------------------------------------
// yg = (scanY + xc * D_skip) * silu(z) ; z = xz[:, :, 512:1024]
// ---------------------------------------------------------------------------
__global__ __launch_bounds__(256) void gate_kernel(
    const float* __restrict__ sy, const float* __restrict__ xc,
    const float* __restrict__ xz, const float* __restrict__ dskip,
    float* __restrict__ yg, int total)
{
    const int idx = blockIdx.x * blockDim.x + threadIdx.x;
    if (idx >= total) return;
    const int d = idx & (DINNER - 1);
    const int g = idx >> 9;
    const float z = xz[(size_t)g * (2 * DINNER) + DINNER + d];
    const float y = sy[idx] + xc[idx] * dskip[d];
    yg[idx] = y * silu_f(z);
}

// ---------------------------------------------------------------------------
extern "C" void kernel_launch(void* const* d_in, const int* in_sizes, int n_in,
                              void* d_out, int out_size, void* d_ws, size_t ws_size,
                              hipStream_t stream) {
    const float* coords   = (const float*)d_in[0];
    const float* emb_W    = (const float*)d_in[1];
    const float* emb_b    = (const float*)d_in[2];
    const float* ln_w     = (const float*)d_in[3];
    const float* ln_b     = (const float*)d_in[4];
    const float* in_W     = (const float*)d_in[5];
    const float* conv_W   = (const float*)d_in[6];
    const float* conv_b   = (const float*)d_in[7];
    const float* xproj_W  = (const float*)d_in[8];
    const float* dtproj_W = (const float*)d_in[9];
    const float* dtproj_b = (const float*)d_in[10];
    const float* A_log    = (const float*)d_in[11];
    const float* D_skip   = (const float*)d_in[12];
    const float* out_W    = (const float*)d_in[13];
    const float* normf_w  = (const float*)d_in[14];
    const float* normf_b  = (const float*)d_in[15];
    const float* head_W   = (const float*)d_in[16];
    float* logits = (float*)d_out;

    // Workspace carve (256B aligned)
    char* ws = (char*)d_ws;
    auto carve = [&](size_t nfloats) -> float* {
        float* p = (float*)ws;
        ws += ((nfloats * sizeof(float) + 255) / 256) * 256;
        return p;
    };
    float* resid = carve((size_t)T_TOK * DMODEL);
    float* xbuf  = carve((size_t)T_TOK * DMODEL);
    float* hbuf  = carve((size_t)T_TOK * DMODEL);
    float* xz    = carve((size_t)T_TOK * 2 * DINNER);
    float* xc    = carve((size_t)T_TOK * DINNER);
    float* dbl   = carve((size_t)T_TOK * 48);
    float* dtb   = carve((size_t)T_TOK * DINNER);
    float* scanY = carve((size_t)T_TOK * DINNER);
    float* yg    = carve((size_t)T_TOK * DINNER);

    const dim3 blk(256);
    const int mBlocks = T_TOK / TM;   // 150, exact

    // 1) embedding
    embed_kernel<<<(T_TOK * DMODEL + 255) / 256, blk, 0, stream>>>(
        coords, emb_W, emb_b, xbuf, T_TOK * DMODEL);

    for (int i = 0; i < NLAYERS; ++i) {
        // 2) residual add + LayerNorm
        add_ln_kernel<<<T_TOK, blk, 0, stream>>>(
            xbuf, resid, ln_w + i * DMODEL, ln_b + i * DMODEL, hbuf, i == 0);

        // 3) in_proj: [T,256] x [1024,256]^T -> xz [T,1024]   (K%32, N%64 ok)
        wmma_gemm_bf16<true, true, false>
            <<<dim3(2 * DINNER / TN, mBlocks), blk, 0, stream>>>(
            hbuf, DMODEL, in_W + (size_t)i * 2 * DINNER * DMODEL, DMODEL,
            xz, 2 * DINNER, nullptr, 2 * DINNER, DMODEL);

        // 4) causal conv + SiLU
        conv_silu_kernel<<<(T_TOK * DINNER + 255) / 256, blk, 0, stream>>>(
            xz, conv_W + (size_t)i * DINNER * 4, conv_b + i * DINNER, xc, T_TOK * DINNER);

        // 5) x_proj: [T,512] x [48,512]^T -> dbl [T,48]       (N=48 ragged)
        wmma_gemm_bf16<true, false, false>
            <<<dim3(1, mBlocks), blk, 0, stream>>>(
            xc, DINNER, xproj_W + (size_t)i * 48 * DINNER, DINNER,
            dbl, 48, nullptr, 48, DINNER);

        // 6) dt_proj (+bias, softplus): [T,16] x [512,16]^T -> dt [T,512]  (K=16 ragged)
        wmma_gemm_bf16<false, true, true>
            <<<dim3(DINNER / TN, mBlocks), blk, 0, stream>>>(
            dbl, 48, dtproj_W + (size_t)i * DINNER * DTRANK, DTRANK,
            dtb, DINNER, dtproj_b + i * DINNER, DINNER, DTRANK);

        // 7) selective scan
        scan_kernel<<<(32 * DINNER + 255) / 256, blk, 0, stream>>>(
            xc, dtb, dbl, A_log + (size_t)i * DINNER * DSTATE, scanY);

        // 8) D-skip + z-gate
        gate_kernel<<<(T_TOK * DINNER + 255) / 256, blk, 0, stream>>>(
            scanY, xc, xz, D_skip + i * DINNER, yg, T_TOK * DINNER);

        // 9) out_proj: [T,512] x [256,512]^T -> xbuf [T,256]
        wmma_gemm_bf16<true, true, false>
            <<<dim3(DMODEL / TN, mBlocks), blk, 0, stream>>>(
            yg, DINNER, out_W + (size_t)i * DMODEL * DINNER, DINNER,
            xbuf, DMODEL, nullptr, DMODEL, DINNER);
    }

    // 10) final add + RMSNorm
    final_rms_kernel<<<T_TOK, blk, 0, stream>>>(xbuf, resid, normf_w, normf_b, hbuf);

    // 11) head: [T,256] x [300,256]^T -> logits [T,300]       (N=300 ragged)
    wmma_gemm_bf16<true, false, false>
        <<<dim3((SEQ + TN - 1) / TN, mBlocks), blk, 0, stream>>>(
        hbuf, DMODEL, head_W, DMODEL, logits, SEQ, nullptr, SEQ, DMODEL);
}
